// LSTMLayer_24524263260903
// MI455X (gfx1250) — compile-verified, matrix operand order checked by reading, not run on hardware
//
#include <hip/hip_runtime.h>
#include <hip/hip_bf16.h>

// ---------------------------------------------------------------------------
// 2-layer LSTM (B=128, T=1024, IN=H=256) + LayerNorm for MI455X (gfx1250).
//   * f16 data / f32-accumulate WMMA (v_wmma_f32_16x16x32_f16)
//   * 8 persistent workgroups (16-row batch tiles) loop over all T;
//     c in VGPRs, h in LDS; fused [x|h] x [Wih;Whh] GEMM (K=512);
//     weights stream from L2 (1 MB/layer, far below 192 MB L2)
//   * x_t tile staged into LDS with global_load_async_to_lds_b128 (ASYNCcnt)
//   * coalesced b128 h writeback from LDS
// ---------------------------------------------------------------------------

typedef _Float16 half_t;
typedef __attribute__((ext_vector_type(16))) _Float16 v16h;
typedef __attribute__((ext_vector_type(8)))  _Float16 v8h;
typedef __attribute__((ext_vector_type(8)))  float    v8f;

#define NUM_LAYERS 2
#define HIDDEN 256
#define IN_SIZE 256
#define BATCH 128
#define TSTEPS 1024
#define GATES (4 * HIDDEN)      // 1024
#define KTOT (IN_SIZE + HIDDEN) // 512
#define LN_EPS 1e-5f

__device__ __forceinline__ float sigmoid_fast(float x) {
  return 1.f / (1.f + __expf(-x));
}
__device__ __forceinline__ float tanh_fast(float x) {
  // tanh(x) = 1 - 2/(exp(2x)+1); v_exp_f32 trans op, co-executes with WMMA
  return 1.f - 2.f / (__expf(2.f * x) + 1.f);
}

// ---------------------------------------------------------------------------
// Pack [W_ih | W_hh] (f32) into a combined f16 matrix Wc[l][g][k], k in [0,512)
// ---------------------------------------------------------------------------
__global__ __launch_bounds__(256) void pack_weights_kernel(
    const float* __restrict__ Wih, const float* __restrict__ Whh,
    half_t* __restrict__ Wc) {
  int idx = blockIdx.x * 256 + threadIdx.x;           // over 2*1024*512
  int l = idx / (GATES * KTOT);
  int g = (idx / KTOT) % GATES;
  int k = idx % KTOT;
  float v;
  if (k < IN_SIZE)
    v = Wih[(size_t)l * GATES * IN_SIZE + (size_t)g * IN_SIZE + k];
  else
    v = Whh[(size_t)l * GATES * HIDDEN + (size_t)g * HIDDEN + (k - IN_SIZE)];
  Wc[(size_t)l * GATES * KTOT + (size_t)g * KTOT + k] = (half_t)v;
}

__global__ __launch_bounds__(256) void pack_bias_kernel(
    const float* __restrict__ bih, const float* __restrict__ bhh,
    float* __restrict__ biasc) {
  int idx = blockIdx.x * 256 + threadIdx.x;           // over 2*1024
  if (idx < NUM_LAYERS * GATES) biasc[idx] = bih[idx] + bhh[idx];
}

// ---------------------------------------------------------------------------
// x [B,T,256] f32  ->  xt [T,B,256] f16 (time-major for the recurrent kernel)
// ---------------------------------------------------------------------------
__global__ __launch_bounds__(256) void transpose_x_kernel(
    const float* __restrict__ x, half_t* __restrict__ xt) {
  size_t idx = (size_t)blockIdx.x * 256 + threadIdx.x; // over T*B*256
  int c = (int)(idx & (IN_SIZE - 1));
  int b = (int)((idx / IN_SIZE) & (BATCH - 1));
  int t = (int)(idx / ((size_t)IN_SIZE * BATCH));
  xt[idx] = (half_t)x[(size_t)b * TSTEPS * IN_SIZE + (size_t)t * IN_SIZE + c];
}

// ---------------------------------------------------------------------------
// One LSTM layer over all T timesteps.
//   in   : [T][B][256] f16      W : [1024][512] f16 ([W_ih|W_hh] rows i,f,g,o)
//   bias : [1024] f32 (b_ih+b_hh)   out : [T][B][256] f16
// grid = 8 blocks (batch tiles of 16), block = 256 threads (8 waves).
// Wave w owns hidden block [w*32, w*32+32): N-tiles nt = gate*2+p at
// column base  gate*256 + w*32 + p*16.
// ---------------------------------------------------------------------------
__global__ __launch_bounds__(256) void lstm_layer_kernel(
    const half_t* __restrict__ in, const half_t* __restrict__ W,
    const float* __restrict__ bias, half_t* __restrict__ out) {
  extern __shared__ char smem[];
  half_t* sAH = (half_t*)smem;  // [16][512] : cols 0..255 = x_t, 256..511 = h

  const int tid  = threadIdx.x;
  const int lane = tid & 31;
  const int w    = tid >> 5;        // wave 0..7
  const int hi   = lane >> 4;       // half-wave select (K/M split)
  const int nn   = lane & 15;       // n within a 16-wide tile
  const int m0   = blockIdx.x * 16; // batch tile origin

  // zero h portion of the A staging tile
  for (int i = tid; i < 16 * HIDDEN; i += 256) {
    int r = i >> 8, j = i & 255;
    sAH[r * KTOT + IN_SIZE + j] = (half_t)0.f;
  }

  // per-lane bias for the 8 owned gate columns
  float biasReg[8];
#pragma unroll
  for (int nt = 0; nt < 8; ++nt) {
    int g0 = (nt >> 1) * HIDDEN + w * 32 + (nt & 1) * 16;
    biasReg[nt] = bias[g0 + nn];
  }

  // cell state in registers: c for (m = r + 8*hi, j = w*32 + p*16 + nn)
  float creg[2][8];
#pragma unroll
  for (int p = 0; p < 2; ++p)
#pragma unroll
    for (int r = 0; r < 8; ++r) creg[p][r] = 0.f;

  // per-thread staging addresses: thread copies 32 bytes of row r
  const int srow = tid >> 4;          // 0..15
  const int sgrp = (tid & 15) * 16;   // 16 halves
  const unsigned ldsXoff =
      (unsigned)(size_t)(void*)(sAH + srow * KTOT + sgrp);

  __syncthreads();

  for (int t = 0; t < TSTEPS; ++t) {
    // ---- stage x_t tile (16 rows x 256 halves) into sAH cols [0,256) ------
    // async DMA: global -> LDS, tracked by ASYNCcnt (no VGPR data movement)
    {
      unsigned long long g =
          (unsigned long long)(in + ((size_t)t * BATCH + m0 + srow) * IN_SIZE +
                               sgrp);
      asm volatile(
          "global_load_async_to_lds_b128 %0, %1, off\n\t"
          "global_load_async_to_lds_b128 %0, %1, off offset:16\n\t"
          "s_wait_asynccnt 0x0"
          :
          : "v"(ldsXoff), "v"(g)
          : "memory");
    }
    // prefetch next timestep's tile into cache (global_prefetch_b8)
    if (lane == 0 && t + 1 < TSTEPS) {
      __builtin_prefetch(in + ((size_t)(t + 1) * BATCH + m0 + w * 2) * IN_SIZE,
                         0, 3);
    }
    __syncthreads();  // x staged + previous h writes visible

    // ---- gates[16 x 32-wide hidden block x 4 gates] = [x|h] @ W^T ---------
    v8f acc[8];
#pragma unroll
    for (int nt = 0; nt < 8; ++nt) acc[nt] = (v8f){0, 0, 0, 0, 0, 0, 0, 0};

#pragma unroll 8
    for (int k0 = 0; k0 < KTOT; k0 += 32) {
      // A fragment (ISA 16-bit A 16x32 layout): row = nn, K = k0+hi*8+{0..7}
      // and k0+hi*8+16+{0..7}
      const half_t* ap = sAH + nn * KTOT + k0 + hi * 8;
      v8h alo = *(const v8h*)(ap);
      v8h ahi = *(const v8h*)(ap + 16);
      v16h aF;
#pragma unroll
      for (int i = 0; i < 8; ++i) { aF[i] = alo[i]; aF[i + 8] = ahi[i]; }

#pragma unroll
      for (int nt = 0; nt < 8; ++nt) {
        int g = (nt >> 1) * HIDDEN + w * 32 + (nt & 1) * 16 + nn;
        const half_t* bp = W + (size_t)g * KTOT + k0 + hi * 8;
        v8h blo = *(const v8h*)(bp);
        v8h bhi = *(const v8h*)(bp + 16);
        v16h bF;
#pragma unroll
        for (int i = 0; i < 8; ++i) { bF[i] = blo[i]; bF[i + 8] = bhi[i]; }
        acc[nt] = __builtin_amdgcn_wmma_f32_16x16x32_f16(
            false, aF, false, bF, (short)0, acc[nt], false, false);
      }
    }
    __syncthreads();  // everyone done reading sAH (old h)

    // ---- pointwise LSTM update, fully register-local per lane -------------
    // element (m = r + 8*hi, j = w*32 + p*16 + nn); gate k in acc[2k+p][r]
#pragma unroll
    for (int p = 0; p < 2; ++p) {
      int j = w * 32 + p * 16 + nn;
#pragma unroll
      for (int r = 0; r < 8; ++r) {
        int m = r + hi * 8;
        float gi = acc[0 + p][r] + biasReg[0 + p];
        float gf = acc[2 + p][r] + biasReg[2 + p];
        float gg = acc[4 + p][r] + biasReg[4 + p];
        float go = acc[6 + p][r] + biasReg[6 + p];
        float i_ = sigmoid_fast(gi);
        float f_ = sigmoid_fast(gf);
        float g_ = tanh_fast(gg);
        float o_ = sigmoid_fast(go);
        float c  = f_ * creg[p][r] + i_ * g_;
        creg[p][r] = c;
        float h = o_ * tanh_fast(c);
        sAH[m * KTOT + IN_SIZE + j] = (half_t)h;  // next-step A matrix
      }
    }
    __syncthreads();  // h(t) complete in LDS

    // ---- coalesced h writeback: LDS -> out as b128 stores -----------------
    {
      const v8h* hsrc = (const v8h*)(sAH + srow * KTOT + IN_SIZE + sgrp);
      v8h h0 = hsrc[0];
      v8h h1 = hsrc[1];
      v8h* dst =
          (v8h*)(out + ((size_t)t * BATCH + m0 + srow) * HIDDEN + sgrp);
      dst[0] = h0;
      dst[1] = h1;
    }
    // next iteration's async x-stage writes cols [0,256) (disjoint from h);
    // the barrier at the top of the next iteration orders everything else.
  }
}

// ---------------------------------------------------------------------------
// LayerNorm over the last timestep of the final layer: [B,256] -> d_out f32
// ---------------------------------------------------------------------------
__global__ __launch_bounds__(256) void layernorm_kernel(
    const half_t* __restrict__ hin, const float* __restrict__ gamma,
    const float* __restrict__ beta, float* __restrict__ outp) {
  __shared__ float red[256];
  int b = blockIdx.x, j = threadIdx.x;
  float v = (float)hin[((size_t)(TSTEPS - 1) * BATCH + b) * HIDDEN + j];
  red[j] = v;
  __syncthreads();
  for (int s = 128; s > 0; s >>= 1) {
    if (j < s) red[j] += red[j + s];
    __syncthreads();
  }
  float mu = red[0] * (1.f / HIDDEN);
  __syncthreads();
  float d = v - mu;
  red[j] = d * d;
  __syncthreads();
  for (int s = 128; s > 0; s >>= 1) {
    if (j < s) red[j] += red[j + s];
    __syncthreads();
  }
  float var = red[0] * (1.f / HIDDEN);
  outp[(size_t)b * HIDDEN + j] = d * rsqrtf(var + LN_EPS) * gamma[j] + beta[j];
}

// ---------------------------------------------------------------------------
extern "C" void kernel_launch(void* const* d_in, const int* in_sizes, int n_in,
                              void* d_out, int out_size, void* d_ws,
                              size_t ws_size, hipStream_t stream) {
  const float* x     = (const float*)d_in[0];
  const float* Wih   = (const float*)d_in[1];
  const float* Whh   = (const float*)d_in[2];
  const float* bih   = (const float*)d_in[3];
  const float* bhh   = (const float*)d_in[4];
  const float* gamma = (const float*)d_in[5];
  const float* beta  = (const float*)d_in[6];
  float* outp = (float*)d_out;

  char* ws = (char*)d_ws;
  // workspace layout
  half_t* Wc    = (half_t*)ws;                                   // 2 MiB
  float*  biasc = (float*)(ws + (size_t)NUM_LAYERS * GATES * KTOT * 2);
  char*   bufs  = ws + (size_t)NUM_LAYERS * GATES * KTOT * 2 + 8192;
  half_t* bufA  = (half_t*)bufs;                                 // 64 MiB
  half_t* bufB  = bufA + (size_t)TSTEPS * BATCH * HIDDEN;        // 64 MiB

  pack_weights_kernel<<<(NUM_LAYERS * GATES * KTOT) / 256, 256, 0, stream>>>(
      Wih, Whh, Wc);
  pack_bias_kernel<<<8, 256, 0, stream>>>(bih, bhh, biasc);
  transpose_x_kernel<<<(int)(((size_t)TSTEPS * BATCH * IN_SIZE) / 256), 256, 0,
                       stream>>>(x, bufA);

  size_t lds = (size_t)16 * KTOT * sizeof(half_t);  // 16 KiB staging tile
  // layer 0: x(f16, time-major) -> bufB
  lstm_layer_kernel<<<BATCH / 16, 256, lds, stream>>>(bufA, Wc, biasc, bufB);
  // layer 1: bufB -> bufA
  lstm_layer_kernel<<<BATCH / 16, 256, lds, stream>>>(
      bufB, Wc + (size_t)GATES * KTOT, biasc + GATES, bufA);

  layernorm_kernel<<<BATCH, 256, 0, stream>>>(bufA, gamma, beta, outp);
}
